// TaskAlignedAssigner_85212151152985
// MI455X (gfx1250) — compile-verified, matrix operand order checked by reading, not run on hardware
//
#include <hip/hip_runtime.h>
#include <hip/hip_bf16.h>

// TaskAlignedAssigner for MI455X (gfx1250, wave32).
// K0a: one-hot(labels) fp16 prebuilt in WMMA A lane layout (393 KB, L2-resident).
// K0b: pd_scores f32->f16, padded C 80->96 (B operand becomes 2x b128 per chunk).
// K1 : align_metric via V_WMMA_F32_16X16X32_F16 (pure load->wmma inner loop) + IoU^6.
// K2 : per-(b,m) 10th-largest threshold (register top-10 + LDS merge).
// K3 : per-(b,n) argmax over m + mask combine + output gathers.

#define B_    32
#define N_    8400
#define M_    64
#define C_    80
#define CP_   96      // C padded to 3 chunks of K=32
#define TOPK_ 10

typedef __attribute__((ext_vector_type(16))) _Float16 v16h;
typedef __attribute__((ext_vector_type(8)))  float    v8f;

union V16H { v16h v; _Float16 e[16]; uint4 q[2]; };
union V8F  { v8f  v; float    e[8];  };
union H8   { _Float16 e[8]; uint4 q; };

// ---------------------------------------------------------------------------
// K0a: A_ws[b][mtile(4)][kc(3)][lane(32)][i(16)] fp16, exact WMMA A layout.
// One block per batch, 384 threads = 4 mtiles x 3 chunks x 32 lanes.
// ---------------------------------------------------------------------------
__global__ void __launch_bounds__(384)
k0a_onehot(const int* __restrict__ gt_labels, _Float16* __restrict__ A_ws) {
    const int b     = blockIdx.x;
    const int t     = threadIdx.x;
    const int lane  = t & 31;
    const int kc    = (t >> 5) % 3;
    const int mtile = (t >> 5) / 3;
    const int hi    = (lane >> 4) & 1;
    const int r     = lane & 15;
    const int lab   = gt_labels[b * M_ + mtile * 16 + r];
    const int kbase = kc * 32;

    V16H a;
    #pragma unroll
    for (int i = 0; i < 16; ++i) {
        // 16-bit A operand K sequence: lanes 0-15 {0..7,16..23}, lanes 16-31 {8..15,24..31}
        const int k = kbase + ((i < 8) ? (hi * 8 + i) : (16 + hi * 8 + (i - 8)));
        a.e[i] = (lab == k) ? (_Float16)1.0f : (_Float16)0.0f;
    }
    _Float16* dst = A_ws + ((((size_t)b * 4 + mtile) * 3 + kc) * 32 + lane) * 16;
    *(v16h*)dst = a.v;   // 32B coalesced store
}

// ---------------------------------------------------------------------------
// K0b: scores_h[b][n][0..95] = fp16(pd_scores[b][n][k]) (0 for k>=80).
// One thread per 8-class group -> one 16B store. grid = (ceil(N*12/256), B).
// ---------------------------------------------------------------------------
__global__ void __launch_bounds__(256)
k0b_cvt(const float* __restrict__ pd_scores, _Float16* __restrict__ scores_h) {
    const int b   = blockIdx.y;
    const int idx = blockIdx.x * 256 + threadIdx.x;     // 0 .. N*12-1
    if (idx >= N_ * (CP_ / 8)) return;
    const int n   = idx / (CP_ / 8);
    const int k0  = (idx % (CP_ / 8)) * 8;

    const float* srow = pd_scores + ((size_t)b * N_ + n) * C_;
    H8 h;
    #pragma unroll
    for (int i = 0; i < 8; ++i) {
        const int k = k0 + i;
        h.e[i] = (k < C_) ? (_Float16)srow[k] : (_Float16)0.0f;
    }
    *(uint4*)(scores_h + ((size_t)b * N_ + n) * CP_ + k0) = h.q;
}

// ---------------------------------------------------------------------------
// K1: grid = (N/16 = 525, B), block = 128 (4 waves = 4 m-tiles of 16).
// Inner loop: 2x b128 (A frag) + 2x b128 (B frag) -> v_wmma, 3 chunks.
// D layout: lane r=lane&15 -> n = n_base+r ; VGPR v -> m = m_base+v+(lane>=16?8:0).
// ---------------------------------------------------------------------------
__global__ void __launch_bounds__(128)
k1_align(const _Float16* __restrict__ scores_h,
         const _Float16* __restrict__ A_ws,
         const float*    __restrict__ pd_bboxes,
         const float*    __restrict__ anc,
         const float*    __restrict__ gt_bboxes,
         float*          __restrict__ alignm) {
    __shared__ float s_box[M_][4];

    const int b   = blockIdx.y;
    const int tid = threadIdx.x;
    const int n_base = blockIdx.x * 16;

    if (tid < M_) {
        #pragma unroll
        for (int j = 0; j < 4; ++j)
            s_box[tid][j] = gt_bboxes[(size_t)(b * M_ + tid) * 4 + j];
    }
    __syncthreads();

    const int wave = tid >> 5;
    const int lane = tid & 31;
    const int hi   = (lane >> 4) & 1;
    const int r    = lane & 15;
    const int m_base = wave * 16;
    const int n      = n_base + r;

    // Keep epilogue data in flight behind the WMMAs (global_prefetch_b8).
    __builtin_prefetch(&pd_bboxes[((size_t)b * N_ + n) * 4], 0, 0);
    __builtin_prefetch(&anc[(size_t)n * 2], 0, 0);

    const _Float16* srow = scores_h + ((size_t)b * N_ + n) * CP_;
    const _Float16* arow = A_ws + (((size_t)b * 4 + wave) * 3) * 32 * 16;

    V8F acc;
    #pragma unroll
    for (int i = 0; i < 8; ++i) acc.e[i] = 0.0f;

    #pragma unroll
    for (int kc = 0; kc < 3; ++kc) {
        V16H a, bm;
        a.v = *(const v16h*)(arow + ((size_t)kc * 32 + lane) * 16);        // 32B
        const int kb = kc * 32 + hi * 8;
        bm.q[0] = *(const uint4*)(srow + kb);        // K {kb..kb+7}
        bm.q[1] = *(const uint4*)(srow + kb + 16);   // K {kb+16..kb+23}
        acc.v = __builtin_amdgcn_wmma_f32_16x16x32_f16(
            /*neg_a=*/false, a.v, /*neg_b=*/false, bm.v,
            /*c_mod=*/(short)0, acc.v, /*reuse_a=*/false, /*reuse_b=*/false);
    }

    // Epilogue: all 8 accumulators share this lane's anchor n.
    const float px1 = pd_bboxes[((size_t)b * N_ + n) * 4 + 0];
    const float py1 = pd_bboxes[((size_t)b * N_ + n) * 4 + 1];
    const float px2 = pd_bboxes[((size_t)b * N_ + n) * 4 + 2];
    const float py2 = pd_bboxes[((size_t)b * N_ + n) * 4 + 3];
    const float ax  = anc[(size_t)n * 2 + 0];
    const float ay  = anc[(size_t)n * 2 + 1];
    const float pa  = (px2 - px1) * (py2 - py1);

    #pragma unroll
    for (int v = 0; v < 8; ++v) {
        const int m = m_base + v + hi * 8;
        const float gx1 = s_box[m][0], gy1 = s_box[m][1];
        const float gx2 = s_box[m][2], gy2 = s_box[m][3];
        const float iw = fmaxf(fminf(px2, gx2) - fmaxf(px1, gx1), 0.0f);
        const float ih = fmaxf(fminf(py2, gy2) - fmaxf(py1, gy1), 0.0f);
        const float inter = iw * ih;
        const float ga  = (gx2 - gx1) * (gy2 - gy1);
        // Fast reciprocal (v_rcp_f32, ~1 ulp): fine for a ranking metric,
        // avoids the 7-op IEEE div_scale/div_fixup expansion per element.
        const float iou = inter * __builtin_amdgcn_rcpf(pa + ga - inter + 1e-7f);
        const float i2 = iou * iou;
        const float i6 = i2 * i2 * i2;
        const float inside =
            (ax >= gx1 && ax <= gx2 && ay >= gy1 && ay <= gy2) ? 1.0f : 0.0f;
        alignm[((size_t)b * M_ + m) * N_ + n] = acc.e[v] * i6 * inside;
    }
}

// ---------------------------------------------------------------------------
// K2: one block per (b,m) row. 10th-largest value -> threshold.
// Merge loops deliberately NOT unrolled: they are a once-per-block serial
// tail; full unroll only bloats static code.
// ---------------------------------------------------------------------------
__device__ __forceinline__ void top10_insert(float* t, float v) {
    if (v > t[TOPK_ - 1]) {
        t[TOPK_ - 1] = v;
        #pragma unroll
        for (int j = TOPK_ - 1; j > 0; --j) {
            if (t[j] > t[j - 1]) { float tmp = t[j]; t[j] = t[j - 1]; t[j - 1] = tmp; }
        }
    }
}

__global__ void __launch_bounds__(256)
k2_thresh(const float* __restrict__ alignm, float* __restrict__ thr) {
    const int bm = blockIdx.x;
    const float* row = alignm + (size_t)bm * N_;

    float t[TOPK_];
    #pragma unroll
    for (int i = 0; i < TOPK_; ++i) t[i] = -1.0f;

    #pragma clang loop unroll(disable)
    for (int nn = threadIdx.x; nn < N_; nn += 256)
        top10_insert(t, row[nn]);

    __shared__ float s[256 * TOPK_];
    #pragma unroll
    for (int i = 0; i < TOPK_; ++i) s[threadIdx.x * TOPK_ + i] = t[i];
    __syncthreads();

    if (threadIdx.x < 32) {
        #pragma clang loop unroll(disable)
        for (int c = 1; c < 8; ++c) {
            const float* o = &s[(threadIdx.x + c * 32) * TOPK_];
            #pragma clang loop unroll(disable)
            for (int i = 0; i < TOPK_; ++i) top10_insert(t, o[i]);
        }
        #pragma unroll
        for (int i = 0; i < TOPK_; ++i) s[threadIdx.x * TOPK_ + i] = t[i];
    }
    __syncthreads();

    if (threadIdx.x == 0) {
        #pragma clang loop unroll(disable)
        for (int c = 1; c < 32; ++c)
            #pragma clang loop unroll(disable)
            for (int i = 0; i < TOPK_; ++i) top10_insert(t, s[c * TOPK_ + i]);
        thr[bm] = t[TOPK_ - 1];
    }
}

// ---------------------------------------------------------------------------
// K3: per anchor: argmax over m (strict > = first-max, matches jnp.argmax),
// positive iff best >= thr[m*] (top-10 membership) and mask_gt[m*] != 0.
// Outputs concat: labels (B*N) | bboxes (B*N*4) | mask_pos (B*N) | gt_idx (B*N i32)
// ---------------------------------------------------------------------------
__global__ void __launch_bounds__(256)
k3_assign(const float* __restrict__ alignm, const float* __restrict__ thr,
          const int* __restrict__ gt_labels, const float* __restrict__ gt_bboxes,
          const float* __restrict__ mask_gt, float* __restrict__ out) {
    __shared__ float s_thr[M_], s_mgt[M_], s_box[M_][4];
    __shared__ int   s_lab[M_];

    const int b = blockIdx.y;
    const int n = blockIdx.x * 256 + threadIdx.x;

    if (threadIdx.x < M_) {
        const int i = b * M_ + threadIdx.x;
        s_thr[threadIdx.x] = thr[i];
        s_mgt[threadIdx.x] = mask_gt[i];
        s_lab[threadIdx.x] = gt_labels[i];
        #pragma unroll
        for (int j = 0; j < 4; ++j) s_box[threadIdx.x][j] = gt_bboxes[(size_t)i * 4 + j];
    }
    __syncthreads();
    if (n >= N_) return;

    float best = -1.0f;
    int   bi   = 0;
    #pragma clang loop unroll_count(8)
    for (int m = 0; m < M_; ++m) {
        const float v = alignm[((size_t)b * M_ + m) * N_ + n];
        if (v > best) { best = v; bi = m; }
    }

    const bool pos = (best >= s_thr[bi]) && (s_mgt[bi] != 0.0f);
    const size_t BN = (size_t)B_ * N_;
    const size_t o  = (size_t)b * N_ + n;

    out[o] = pos ? (float)s_lab[bi] : 0.0f;                        // target_labels
    #pragma unroll
    for (int j = 0; j < 4; ++j)                                    // target_bboxes
        out[BN + o * 4 + j] = pos ? s_box[bi][j] : 0.0f;
    out[BN * 5 + o] = pos ? 1.0f : 0.0f;                           // mask_pos
    ((int*)out)[BN * 6 + o] = pos ? bi : 0;                        // target_gt_idx (i32)
}

// ---------------------------------------------------------------------------
extern "C" void kernel_launch(void* const* d_in, const int* in_sizes, int n_in,
                              void* d_out, int out_size, void* d_ws, size_t ws_size,
                              hipStream_t stream) {
    const float* pd_scores = (const float*)d_in[0];   // (B,N,C)
    const float* pd_bboxes = (const float*)d_in[1];   // (B,N,4)
    const float* anc       = (const float*)d_in[2];   // (N,2)
    const int*   gt_labels = (const int*)  d_in[3];   // (B,M,1)
    const float* gt_bboxes = (const float*)d_in[4];   // (B,M,4)
    const float* mask_gt   = (const float*)d_in[5];   // (B,M,1)

    // Workspace layout (all 16B-aligned):
    float*    alignm   = (float*)d_ws;                               // B*M*N f32 (68.8 MB)
    _Float16* scores_h = (_Float16*)(alignm + (size_t)B_ * M_ * N_); // B*N*96 f16 (51.6 MB)
    _Float16* A_ws     = scores_h + (size_t)B_ * N_ * CP_;           // B*4*3*32*16 f16 (393 KB)
    float*    thr      = (float*)(A_ws + (size_t)B_ * 4 * 3 * 32 * 16); // B*M f32

    k0a_onehot<<<B_, 384, 0, stream>>>(gt_labels, A_ws);

    dim3 g0b((N_ * (CP_ / 8) + 255) / 256, B_);
    k0b_cvt<<<g0b, 256, 0, stream>>>(pd_scores, scores_h);

    dim3 g1(N_ / 16, B_);                                 // 525 x 32, exact (no WMMA tail)
    k1_align<<<g1, 128, 0, stream>>>(scores_h, A_ws, pd_bboxes, anc, gt_bboxes, alignm);

    k2_thresh<<<B_ * M_, 256, 0, stream>>>(alignm, thr);

    dim3 g3((N_ + 255) / 256, B_);
    k3_assign<<<g3, 256, 0, stream>>>(alignm, thr, gt_labels, gt_bboxes, mask_gt, (float*)d_out);
}
